// transformer_env_ARC_52415780880424
// MI455X (gfx1250) — compile-verified, matrix operand order checked by reading, not run on hardware
//
#include <hip/hip_runtime.h>
#include <hip/hip_bf16.h>

// Problem constants (from the reference)
#define IN_ 64
#define D_  256
#define NH_ 4
#define HD_ 64
#define L_  2
#define FF_ 2048
#define B_  16
#define T_  64
#define W_  64   // HIST + 1

typedef __bf16 bf16;
typedef __attribute__((ext_vector_type(16))) __bf16 v16bf;
typedef __attribute__((ext_vector_type(8)))  __bf16 v8bf;
typedef __attribute__((ext_vector_type(8)))  float  v8f;

union Tile16x32 { v16bf v; v8bf h[2]; };

// Load a 16x32 bf16 WMMA operand tile from row-major storage.
// `base` points at element (row0, k0); `ld` = row stride in elements.
// ISA 7.12.2 16-bit A/B layout: lane L owns row (L&15); lanes 0-15 carry
// K = {0..7, 16..23}, lanes 16-31 carry K = {8..15, 24..31}.
__device__ __forceinline__ v16bf load_tile(const bf16* base, int ld, int lane) {
  const int r  = lane & 15;
  const int hh = (lane >> 4) * 8;
  const bf16* p = base + r * ld;
  Tile16x32 u;
  u.h[0] = *(const v8bf*)(p + hh);
  u.h[1] = *(const v8bf*)(p + 16 + hh);
  return u.v;
}

__device__ __forceinline__ v8f wmma_bf16(v16bf a, v16bf b, v8f c) {
  // 8 args: (neg_a, A, neg_b, B, c_mod, C, reuse_a, reuse_b)
  return __builtin_amdgcn_wmma_f32_16x16x32_bf16(false, a, false, b, (short)0, c,
                                                 false, false);
}

#define ZERO8 {0.f,0.f,0.f,0.f,0.f,0.f,0.f,0.f}

// ---------------------------------------------------------------------------
// fp32 -> bf16 cast for weights
__global__ void cvt_f32_bf16(const float* __restrict__ s, bf16* __restrict__ d, int n) {
  int i = blockIdx.x * 256 + threadIdx.x;
  if (i < n) d[i] = (bf16)s[i];
}

// ---------------------------------------------------------------------------
// h = relu(x @ fc_w^T + fc_b)  — IN=64 dot products, bandwidth-trivial
__global__ __launch_bounds__(256) void fc_relu(const float* __restrict__ x,
                                               const float* __restrict__ w,
                                               const float* __restrict__ b,
                                               float* __restrict__ h) {
  __shared__ float xr[IN_];
  const int row = blockIdx.x;                 // 0 .. B*T-1
  if (threadIdx.x < IN_) xr[threadIdx.x] = x[row * IN_ + threadIdx.x];
  __syncthreads();
  const int n = threadIdx.x;                  // output feature
  float acc = b[n];
  #pragma unroll 8
  for (int i = 0; i < IN_; ++i) acc += xr[i] * w[n * IN_ + i];
  h[row * D_ + n] = fmaxf(acc, 0.f);
}

// ---------------------------------------------------------------------------
// Fully fused per-window transformer: one workgroup per (b,t).
// Dynamic LDS layout (170 KB):
//   win  f32  64x256  @0       65536 B   residual stream (stays resident)
//   abuf bf16 64x256  @65536   32768 B   bf16 copy of win (A operand)
//   cbuf bf16 64x256  @98304   32768 B   attn ctx / FF hidden chunk
//   qb   bf16 64x64   @131072   8192 B   q (then softmax probs)
//   kb   bf16 64x64   @139264   8192 B   k
//   vb   bf16 64x64   @147456   8192 B   v stored TRANSPOSED (feature-major)
//   sbuf f32  64x64   @155648  16384 B   attention scores
//   red  f32  512     @172032   2048 B   softmax/LN reduction scratch
__global__ __launch_bounds__(256, 1)
void fused_window_transformer(const float* __restrict__ g_h,
                              const bf16*  __restrict__ w_qkv,  // (L,768,256) bf16
                              const float* __restrict__ b_qkv,  // (L,768)
                              const bf16*  __restrict__ w_out,  // (L,256,256)
                              const float* __restrict__ b_out,  // (L,256)
                              const float* __restrict__ ln1w,
                              const float* __restrict__ ln1b,
                              const bf16*  __restrict__ w_ff1,  // (L,2048,256)
                              const float* __restrict__ b_ff1,  // (L,2048)
                              const bf16*  __restrict__ w_ff2,  // (L,256,2048)
                              const float* __restrict__ b_ff2,  // (L,256)
                              const float* __restrict__ ln2w,
                              const float* __restrict__ ln2b,
                              float* __restrict__ g_last) {
  extern __shared__ char smem[];
  float* win  = (float*)smem;
  bf16*  abuf = (bf16*)(smem + 65536);
  bf16*  cbuf = (bf16*)(smem + 98304);
  bf16*  qb   = (bf16*)(smem + 131072);
  bf16*  kb   = (bf16*)(smem + 139264);
  bf16*  vb   = (bf16*)(smem + 147456);
  float* sbuf = (float*)(smem + 155648);
  float* red  = (float*)(smem + 172032);

  const int tid  = threadIdx.x;
  const int lane = tid & 31;
  const int wv   = tid >> 5;              // wave id 0..7
  const int cn   = lane & 15;             // N within a 16x16 C tile
  const int rb   = (lane >> 4) * 8;       // M base within a 16x16 C tile

  const int bt = blockIdx.x;
  const int b  = bt / T_;
  const int t  = bt % T_;

  // row / quarter assignment for row-parallel LN & softmax (4 threads per row)
  const int rr = tid >> 2;                // row 0..63
  const int rq = tid & 3;                 // quarter 0..3

  // -------- gather causal window: win[r,:] = h[b, clamp(t-63+r, 0), :]
  for (int idx = tid; idx < W_ * D_; idx += 256) {
    int r = idx >> 8, c = idx & 255;
    int sr = t - (W_ - 1) + r; if (sr < 0) sr = 0;
    float v = g_h[(b * T_ + sr) * D_ + c];
    win[idx]  = v;
    abuf[idx] = (bf16)v;
  }
  __syncthreads();

  for (int l = 0; l < L_; ++l) {
    const bf16* Wqkv = w_qkv + (size_t)l * 768 * D_;
    // L2 prefetch hint for the upcoming FF weight stream (global_prefetch_b8)
    if (lane == 0) __builtin_prefetch(w_ff1 + (size_t)l * FF_ * D_, 0, 1);

    // ================= attention (per head) =================
    for (int hd = 0; hd < NH_; ++hd) {
      // --- q,k,v head projections: 3 x (64x64) = 48 tiles, 6 per wave
      #pragma unroll
      for (int tt = 0; tt < 6; ++tt) {
        int tile  = wv * 6 + tt;
        int which = tile >> 4;            // 0=q 1=k 2=v
        int tm    = (tile & 15) >> 2;
        int tn    = tile & 3;
        int wrow  = which * D_ + hd * HD_ + tn * 16;
        // dual accumulators: independent WMMA chains fill hazard windows
        v8f a0 = ZERO8, a1 = ZERO8;
        #pragma unroll
        for (int kk = 0; kk < 8; kk += 2) {
          a0 = wmma_bf16(load_tile(abuf + (tm * 16) * D_ + kk * 32, D_, lane),
                         load_tile(Wqkv + (size_t)wrow * D_ + kk * 32, D_, lane), a0);
          a1 = wmma_bf16(load_tile(abuf + (tm * 16) * D_ + (kk + 1) * 32, D_, lane),
                         load_tile(Wqkv + (size_t)wrow * D_ + (kk + 1) * 32, D_, lane), a1);
        }
        v8f acc = a0 + a1;
        float bias = b_qkv[l * 768 + wrow + cn];
        #pragma unroll
        for (int i = 0; i < 8; ++i) {
          int m = tm * 16 + rb + i, n = tn * 16 + cn;
          float v = acc[i] + bias;
          if (which == 2) vb[n * W_ + m] = (bf16)v;      // V transposed
          else if (which == 1) kb[m * HD_ + n] = (bf16)v;
          else qb[m * HD_ + n] = (bf16)v;
        }
      }
      __syncthreads();

      // --- scores s = q k^T * 1/8 + causal-window mask: 16 tiles, 2 per wave
      #pragma unroll
      for (int tt = 0; tt < 2; ++tt) {
        int tile = wv * 2 + tt;
        int tm = tile >> 2, tn = tile & 3;
        v8f a0 = ZERO8, a1 = ZERO8;
        a0 = wmma_bf16(load_tile(qb + (tm * 16) * HD_, HD_, lane),
                       load_tile(kb + (tn * 16) * HD_, HD_, lane), a0);
        a1 = wmma_bf16(load_tile(qb + (tm * 16) * HD_ + 32, HD_, lane),
                       load_tile(kb + (tn * 16) * HD_ + 32, HD_, lane), a1);
        v8f acc = a0 + a1;
        int n = tn * 16 + cn;                             // key index
        float mask = (n >= (W_ - 1) - t) ? 0.f : -1e30f;  // j = t-63+n >= 0
        #pragma unroll
        for (int i = 0; i < 8; ++i)
          sbuf[(tm * 16 + rb + i) * W_ + n] = acc[i] * 0.125f + mask;
      }
      __syncthreads();

      // --- softmax rows (4 threads per row); probs (bf16) overwrite qb
      {
        const int cbase = rq * 16;
        float vals[16];
        float mx = -3.4e38f;
        #pragma unroll
        for (int i = 0; i < 16; ++i) {
          vals[i] = sbuf[rr * W_ + cbase + i];
          mx = fmaxf(mx, vals[i]);
        }
        red[tid] = mx;
        __syncthreads();
        mx = fmaxf(fmaxf(red[rr * 4], red[rr * 4 + 1]),
                   fmaxf(red[rr * 4 + 2], red[rr * 4 + 3]));
        float s = 0.f;
        #pragma unroll
        for (int i = 0; i < 16; ++i) { vals[i] = __expf(vals[i] - mx); s += vals[i]; }
        red[256 + tid] = s;
        __syncthreads();
        s = red[256 + rr * 4] + red[256 + rr * 4 + 1] +
            red[256 + rr * 4 + 2] + red[256 + rr * 4 + 3];
        float inv = 1.f / s;
        #pragma unroll
        for (int i = 0; i < 16; ++i)
          qb[rr * W_ + cbase + i] = (bf16)(vals[i] * inv);
      }
      __syncthreads();

      // --- ctx_h = P @ V : 16 tiles, 2 per wave -> cbuf[:, hd*64 + n]
      #pragma unroll
      for (int tt = 0; tt < 2; ++tt) {
        int tile = wv * 2 + tt;
        int tm = tile >> 2, tn = tile & 3;
        v8f a0 = ZERO8, a1 = ZERO8;
        a0 = wmma_bf16(load_tile(qb + (tm * 16) * W_, W_, lane),
                       load_tile(vb + (tn * 16) * W_, W_, lane), a0);
        a1 = wmma_bf16(load_tile(qb + (tm * 16) * W_ + 32, W_, lane),
                       load_tile(vb + (tn * 16) * W_ + 32, W_, lane), a1);
        v8f acc = a0 + a1;
        #pragma unroll
        for (int i = 0; i < 8; ++i)
          cbuf[(tm * 16 + rb + i) * D_ + hd * HD_ + tn * 16 + cn] = (bf16)acc[i];
      }
      __syncthreads();
    } // heads

    // ================= out projection + residual =================
    // 64x256 output = 64 tiles, 8 per wave; accumulate into win (f32)
    #pragma unroll
    for (int tt = 0; tt < 8; ++tt) {
      int tile = wv * 8 + tt;
      int tm = tile >> 4, tn = tile & 15;
      v8f a0 = ZERO8, a1 = ZERO8;
      #pragma unroll
      for (int kk = 0; kk < 8; kk += 2) {
        a0 = wmma_bf16(load_tile(cbuf + (tm * 16) * D_ + kk * 32, D_, lane),
                       load_tile(w_out + ((size_t)l * D_ + tn * 16) * D_ + kk * 32, D_, lane), a0);
        a1 = wmma_bf16(load_tile(cbuf + (tm * 16) * D_ + (kk + 1) * 32, D_, lane),
                       load_tile(w_out + ((size_t)l * D_ + tn * 16) * D_ + (kk + 1) * 32, D_, lane), a1);
      }
      v8f acc = a0 + a1;
      float bias = b_out[l * D_ + tn * 16 + cn];
      #pragma unroll
      for (int i = 0; i < 8; ++i) {
        int m = tm * 16 + rb + i, n = tn * 16 + cn;
        win[m * D_ + n] += acc[i] + bias;
      }
    }
    __syncthreads();

    // ================= LN1 (refresh abuf), 4 threads per row ============
    {
      float s = 0.f, s2 = 0.f;
      for (int c = rq * 64; c < rq * 64 + 64; ++c) {
        float v = win[rr * D_ + c]; s += v; s2 += v * v;
      }
      red[tid] = s; red[256 + tid] = s2;
      __syncthreads();
      float mean = (red[rr * 4] + red[rr * 4 + 1] + red[rr * 4 + 2] + red[rr * 4 + 3]) * (1.f / D_);
      float ex2  = (red[256 + rr * 4] + red[256 + rr * 4 + 1] +
                    red[256 + rr * 4 + 2] + red[256 + rr * 4 + 3]) * (1.f / D_);
      float inv = rsqrtf(ex2 - mean * mean + 1e-5f);
      for (int c = rq * 64; c < rq * 64 + 64; ++c) {
        float y = (win[rr * D_ + c] - mean) * inv * ln1w[l * D_ + c] + ln1b[l * D_ + c];
        win[rr * D_ + c]  = y;
        abuf[rr * D_ + c] = (bf16)y;
      }
    }
    __syncthreads();

    // ================= FF: relu(win@W1^T)@W2^T, chunked over FF =========
    // Each wave holds 8 persistent f32 accumulator tiles of the 64x256 output
    // (columns tn = wv*2, wv*2+1; all 4 row tiles).
    v8f facc[8];
    #pragma unroll
    for (int i = 0; i < 8; ++i) facc[i] = (v8f)ZERO8;

    for (int fb = 0; fb < FF_ / 128; ++fb) {
      // hidden chunk: 64x128 = 32 tiles, 4 per wave -> cbuf (ld 128)
      #pragma unroll
      for (int tt = 0; tt < 4; ++tt) {
        int tile = wv * 4 + tt;
        int tm = tile >> 3, tn = tile & 7;
        v8f a0 = ZERO8, a1 = ZERO8;
        #pragma unroll
        for (int kk = 0; kk < 8; kk += 2) {
          a0 = wmma_bf16(load_tile(abuf + (tm * 16) * D_ + kk * 32, D_, lane),
                         load_tile(w_ff1 + ((size_t)l * FF_ + fb * 128 + tn * 16) * D_ + kk * 32, D_, lane), a0);
          a1 = wmma_bf16(load_tile(abuf + (tm * 16) * D_ + (kk + 1) * 32, D_, lane),
                         load_tile(w_ff1 + ((size_t)l * FF_ + fb * 128 + tn * 16) * D_ + (kk + 1) * 32, D_, lane), a1);
        }
        v8f acc = a0 + a1;
        float bias = b_ff1[l * FF_ + fb * 128 + tn * 16 + cn];
        #pragma unroll
        for (int i = 0; i < 8; ++i)
          cbuf[(tm * 16 + rb + i) * 128 + tn * 16 + cn] = (bf16)fmaxf(acc[i] + bias, 0.f);
      }
      __syncthreads();
      // accumulate chunk contribution into persistent output tiles
      // (8 independent accumulator chains -> scheduler interleaves WMMAs)
      #pragma unroll
      for (int tt = 0; tt < 8; ++tt) {
        int tm = tt >> 1, tn = wv * 2 + (tt & 1);
        #pragma unroll
        for (int kk = 0; kk < 4; ++kk)
          facc[tt] = wmma_bf16(load_tile(cbuf + (tm * 16) * 128 + kk * 32, 128, lane),
                               load_tile(w_ff2 + ((size_t)l * D_ + tn * 16) * FF_ + fb * 128 + kk * 32, FF_, lane),
                               facc[tt]);
      }
      __syncthreads();
    }

    // residual add + ff2 bias
    #pragma unroll
    for (int tt = 0; tt < 8; ++tt) {
      int tm = tt >> 1, tn = wv * 2 + (tt & 1);
      float bias = b_ff2[l * D_ + tn * 16 + cn];
      #pragma unroll
      for (int i = 0; i < 8; ++i) {
        int m = tm * 16 + rb + i, n = tn * 16 + cn;
        win[m * D_ + n] += facc[tt][i] + bias;
      }
    }
    __syncthreads();

    // ================= LN2 (refresh abuf), 4 threads per row ============
    {
      float s = 0.f, s2 = 0.f;
      for (int c = rq * 64; c < rq * 64 + 64; ++c) {
        float v = win[rr * D_ + c]; s += v; s2 += v * v;
      }
      red[tid] = s; red[256 + tid] = s2;
      __syncthreads();
      float mean = (red[rr * 4] + red[rr * 4 + 1] + red[rr * 4 + 2] + red[rr * 4 + 3]) * (1.f / D_);
      float ex2  = (red[256 + rr * 4] + red[256 + rr * 4 + 1] +
                    red[256 + rr * 4 + 2] + red[256 + rr * 4 + 3]) * (1.f / D_);
      float inv = rsqrtf(ex2 - mean * mean + 1e-5f);
      for (int c = rq * 64; c < rq * 64 + 64; ++c) {
        float y = (win[rr * D_ + c] - mean) * inv * ln2w[l * D_ + c] + ln2b[l * D_ + c];
        win[rr * D_ + c]  = y;
        abuf[rr * D_ + c] = (bf16)y;
      }
    }
    __syncthreads();
  } // layers

  // emit last window row (k = W-1)
  if (tid < D_) g_last[(size_t)bt * D_ + tid] = win[(W_ - 1) * D_ + tid];
}

// ---------------------------------------------------------------------------
// classifier: logits = relu(last@c1^T+b1)@c2^T+b2 ; log_softmax over 2
__global__ __launch_bounds__(128) void classifier_head(const float* __restrict__ last,
                                                       const float* __restrict__ w1,
                                                       const float* __restrict__ b1,
                                                       const float* __restrict__ w2,
                                                       const float* __restrict__ b2,
                                                       float* __restrict__ out) {
  __shared__ float row[D_];
  __shared__ float hc[D_ / 2];
  __shared__ float lg[2];
  const int bt = blockIdx.x;
  for (int i = threadIdx.x; i < D_; i += 128) row[i] = last[(size_t)bt * D_ + i];
  __syncthreads();
  float acc = b1[threadIdx.x];
  #pragma unroll 8
  for (int i = 0; i < D_; ++i) acc += row[i] * w1[threadIdx.x * D_ + i];
  hc[threadIdx.x] = fmaxf(acc, 0.f);
  __syncthreads();
  if (threadIdx.x < 2) {
    float a2 = b2[threadIdx.x];
    for (int i = 0; i < D_ / 2; ++i) a2 += hc[i] * w2[threadIdx.x * (D_ / 2) + i];
    lg[threadIdx.x] = a2;
  }
  __syncthreads();
  if (threadIdx.x == 0) {
    float m = fmaxf(lg[0], lg[1]);
    float lse = m + logf(__expf(lg[0] - m) + __expf(lg[1] - m));
    out[bt * 2 + 0] = lg[0] - lse;
    out[bt * 2 + 1] = lg[1] - lse;
  }
}

// ---------------------------------------------------------------------------
extern "C" void kernel_launch(void* const* d_in, const int* in_sizes, int n_in,
                              void* d_out, int out_size, void* d_ws, size_t ws_size,
                              hipStream_t stream) {
  (void)in_sizes; (void)n_in; (void)out_size; (void)ws_size;
  const float* x     = (const float*)d_in[0];
  const float* fc_w  = (const float*)d_in[1];
  const float* fc_b  = (const float*)d_in[2];
  const float* qkv_w = (const float*)d_in[3];
  const float* qkv_b = (const float*)d_in[4];
  const float* out_w = (const float*)d_in[5];
  const float* out_b = (const float*)d_in[6];
  const float* ln1_w = (const float*)d_in[7];
  const float* ln1_b = (const float*)d_in[8];
  const float* ff1_w = (const float*)d_in[9];
  const float* ff1_b = (const float*)d_in[10];
  const float* ff2_w = (const float*)d_in[11];
  const float* ff2_b = (const float*)d_in[12];
  const float* ln2_w = (const float*)d_in[13];
  const float* ln2_b = (const float*)d_in[14];
  const float* c1w   = (const float*)d_in[15];
  const float* c1b   = (const float*)d_in[16];
  const float* c2w   = (const float*)d_in[17];
  const float* c2b   = (const float*)d_in[18];

  char* ws = (char*)d_ws;
  size_t off = 0;
  float* g_h     = (float*)(ws + off); off += (size_t)B_ * T_ * D_ * 4;      // 1 MB
  bf16*  qkvw_bf = (bf16*) (ws + off); off += (size_t)L_ * 3 * D_ * D_ * 2;  // 768 KB
  bf16*  outw_bf = (bf16*) (ws + off); off += (size_t)L_ * D_ * D_ * 2;      // 256 KB
  bf16*  ff1w_bf = (bf16*) (ws + off); off += (size_t)L_ * FF_ * D_ * 2;     // 2 MB
  bf16*  ff2w_bf = (bf16*) (ws + off); off += (size_t)L_ * D_ * FF_ * 2;     // 2 MB
  float* g_last  = (float*)(ws + off); off += (size_t)B_ * T_ * D_ * 4;      // 1 MB

  // weights -> bf16 (resident in L2 for the whole dispatch)
  int n;
  n = L_ * 3 * D_ * D_; cvt_f32_bf16<<<(n + 255) / 256, 256, 0, stream>>>(qkv_w, qkvw_bf, n);
  n = L_ * D_ * D_;     cvt_f32_bf16<<<(n + 255) / 256, 256, 0, stream>>>(out_w, outw_bf, n);
  n = L_ * FF_ * D_;    cvt_f32_bf16<<<(n + 255) / 256, 256, 0, stream>>>(ff1_w, ff1w_bf, n);
  n = L_ * D_ * FF_;    cvt_f32_bf16<<<(n + 255) / 256, 256, 0, stream>>>(ff2_w, ff2w_bf, n);

  // front FC
  fc_relu<<<B_ * T_, 256, 0, stream>>>(x, fc_w, fc_b, g_h);

  // fused per-window transformer (170 KB dynamic LDS, 1 WG per WGP)
  const size_t smem_bytes = 174080;
  fused_window_transformer<<<B_ * T_, 256, smem_bytes, stream>>>(
      g_h, qkvw_bf, qkv_b, outw_bf, out_b, ln1_w, ln1_b,
      ff1w_bf, ff1_b, ff2w_bf, ff2_b, ln2_w, ln2_b, g_last);

  // classifier + log_softmax
  classifier_head<<<B_ * T_, 128, 0, stream>>>(g_last, c1w, c1b, c2w, c2b, (float*)d_out);
}
